// RGCNModel_38388417692043
// MI455X (gfx1250) — matrix operand order repaired and checked by reading, not compiled, not run on hardware
//
#include <hip/hip_runtime.h>

// ---------------------------------------------------------------------------
// RGCN forward for MI455X (gfx1250, wave32, WMMA bf16 16x16x32).
//
// Layer 1: aggregate-first.  A1[d, r, :] = sum_{e:(d,r)} x[src]*w[d,r]  (64-wide)
//          x1 = relu( [A1 | emb] @ [W1_flat ; root1] + b1 )   GEMM [N,1088]x[1088,128]
// Layer 2: transform-first. H2 = x1 @ [W2_0 .. W2_15]         GEMM [N,128]x[128,1024]
//          out = x1 @ root2 + b2  (+ per-edge scatter of H2 messages * w)
//
// B matrices are pre-packed once per launch into fragment-ready bf16 tiles:
//   Bpk[ct][kt][lane][t]  (32x16 tile per (kt,ct), 32B contiguous per lane)
// gemm1 double-buffers its A tile through CDNA5 async global->LDS DMA,
// overlapping the DMA of tile kt+1 with the WMMA of tile kt.
// ---------------------------------------------------------------------------

#define NODES 50000
#define RR    16
#define EMBD  64
#define HIDD  128
#define NEDGE 1600000
#define KT1   34            // K tiles in layer-1 GEMM: 1024/32 (W1) + 2 (root1)

typedef __attribute__((ext_vector_type(16))) __bf16 v16bf;
typedef __attribute__((ext_vector_type(2)))  __bf16 v2bf;
typedef __attribute__((ext_vector_type(8)))  float  v8f;

__device__ __forceinline__ __bf16 f2bf(float f) { return (__bf16)f; }

__device__ __forceinline__ unsigned lds_off(const void* p) {
    // generic AS pointer to LDS: low 32 bits are the wave-relative LDS offset
    return (unsigned)(uintptr_t)p;
}

__device__ __forceinline__ void async_load_b128(unsigned laddr, const void* gaddr) {
    asm volatile("global_load_async_to_lds_b128 %0, %1, off"
                 :: "v"(laddr), "v"((unsigned long long)gaddr) : "memory");
}

// ---------------------------------------------------------------- utilities

__global__ void fill_zero4_kernel(float4* __restrict__ p, int n4) {
    int i = blockIdx.x * blockDim.x + threadIdx.x;
    if (i < n4) p[i] = make_float4(0.f, 0.f, 0.f, 0.f);
}

__global__ void count_kernel(const int* __restrict__ edge_index,
                             const int* __restrict__ edge_type,
                             float* __restrict__ cnt) {
    int e = blockIdx.x * blockDim.x + threadIdx.x;
    if (e < NEDGE) {
        int d = edge_index[NEDGE + e];
        int r = edge_type[e];
        atomicAdd(&cnt[d * RR + r], 1.0f);
    }
}

__global__ void invert_kernel(float* __restrict__ w, int n) {
    int i = blockIdx.x * blockDim.x + threadIdx.x;
    if (i < n) w[i] = 1.0f / fmaxf(w[i], 1.0f);
}

// ------------------------------------------------- B-matrix fragment packing
// Source element (k, n) lives at src[k*sK + (n>>6)*sR + (n&63)]; destination
// is fragment order: dst[((ct*KTtot + ktBase + kt)*32 + lane)*16 + t] where
// lane half h, element t holds K = 8h + t (t<8) or 16 + 8h + (t-8) (t>=8).

__global__ void pack_b_kernel(const float* __restrict__ src,
                              __bf16* __restrict__ dst,
                              int Ksrc, int Ncols, int sK, int sR,
                              int KTtot, int ktBase, int ktLog2) {
    int tid = blockIdx.x * blockDim.x + threadIdx.x;
    if (tid >= Ksrc * Ncols) return;
    int t    = tid & 15;
    int lane = (tid >> 4) & 31;
    int rest = tid >> 9;
    int kt = rest & ((1 << ktLog2) - 1);
    int ct = rest >> ktLog2;
    int k  = (kt << 5) + ((t >> 3) << 4) + ((lane >> 4) << 3) + (t & 7);
    int n  = (ct << 4) + (lane & 15);
    dst[(((size_t)ct * KTtot + ktBase + kt) * 32 + lane) * 16 + t] =
        f2bf(src[(size_t)k * sK + (n >> 6) * sR + (n & 63)]);
}

// --------------------------------------------------------- edge scatter ops
// One wave (32 lanes) per edge; each lane moves a float2 (64 floats total).

__global__ __launch_bounds__(256) void scatter1_kernel(
    const int* __restrict__ edge_index, const int* __restrict__ edge_type,
    const float* __restrict__ emb, const float* __restrict__ wseg,
    float* __restrict__ A1) {
    int widx = (int)((blockIdx.x * blockDim.x + threadIdx.x) >> 5);
    int lane = threadIdx.x & 31;
    if (widx >= NEDGE) return;
    int s = edge_index[widx];
    int d = edge_index[NEDGE + widx];
    int r = edge_type[widx];
    float wv = wseg[d * RR + r];
    float2 xs = ((const float2*)(emb + (size_t)s * EMBD))[lane];
    float* dst = A1 + (size_t)d * (RR * EMBD) + r * EMBD + 2 * lane;
    atomicAdd(dst,     xs.x * wv);
    atomicAdd(dst + 1, xs.y * wv);
}

__global__ __launch_bounds__(256) void scatter2_kernel(
    const int* __restrict__ edge_index, const int* __restrict__ edge_type,
    const float* __restrict__ H2, const float* __restrict__ wseg,
    float* __restrict__ out) {
    int widx = (int)((blockIdx.x * blockDim.x + threadIdx.x) >> 5);
    int lane = threadIdx.x & 31;
    if (widx >= NEDGE) return;
    int s = edge_index[widx];
    int d = edge_index[NEDGE + widx];
    int r = edge_type[widx];
    float wv = wseg[d * RR + r];
    float2 hs = ((const float2*)(H2 + (size_t)s * (RR * EMBD) + r * EMBD))[lane];
    float* o = out + (size_t)d * EMBD + 2 * lane;
    atomicAdd(o,     hs.x * wv);
    atomicAdd(o + 1, hs.y * wv);
}

// --------------------------------------------------------------- GEMM 1
// x1 = relu([A1|emb] @ B1pk + b1), K = 1088 (34 tiles). 8 waves/block; each
// wave owns one 16-col tile. A tile double-buffered via async global->LDS
// DMA: tile kt+1 is in flight while tile kt feeds the WMMA pipe.

__global__ __launch_bounds__(256) void gemm1_kernel(
    const float* __restrict__ A1,     // [N, 1024]
    const float* __restrict__ emb,    // [N, 64]
    const __bf16* __restrict__ B1pk,  // [8][34][32][16]
    const float* __restrict__ b1,     // [128]
    float* __restrict__ x1) {         // [N, 128]
    __shared__ float As[2][16 * 32];  // double-buffered fp32 DMA target (4 KB)
    int row0 = blockIdx.x * 16;
    int wave = threadIdx.x >> 5;
    int lane = threadIdx.x & 31;
    int tid  = threadIdx.x;

    // per-lane DMA descriptors (128 issuing lanes cover the 16x32 tile)
    bool issuer = tid < 128;
    int rI = tid >> 3;                 // tile row 0..15
    int qI = tid & 7;                  // 4-float chunk 0..7
    const float* gA = A1  + (size_t)(row0 + rI) * (RR * EMBD) + qI * 4;
    const float* gE = emb + (size_t)(row0 + rI) * EMBD + qI * 4;
    unsigned lofs = (unsigned)((rI * 32 + qI * 4) * 4);
    unsigned lbuf0 = lds_off(&As[0][0]) + lofs;
    unsigned lbuf1 = lds_off(&As[1][0]) + lofs;

    const __bf16* bptr = B1pk + ((size_t)wave * KT1) * 512 + lane * 16;

    if (issuer) async_load_b128(lbuf0, gA);          // prologue: tile 0

    v8f acc = {};
    for (int kt = 0; kt < KT1; ++kt) {
        if (kt + 1 < KT1) {
            if (issuer) {                            // prefetch tile kt+1
                int kn = kt + 1;
                const float* g = (kn < 32) ? (gA + kn * 32) : (gE + (kn - 32) * 32);
                async_load_b128((kn & 1) ? lbuf1 : lbuf0, g);
            }
            asm volatile("s_wait_asynccnt 0x1" ::: "memory");  // tile kt landed
        } else {
            asm volatile("s_wait_asynccnt 0x0" ::: "memory");
        }
        __syncthreads();                             // tile kt visible to all

        const float* Asrc = As[kt & 1];
        v16bf afrag;
        int m  = lane & 15;
        int kb = (lane >> 4) * 8;
#pragma unroll
        for (int t = 0; t < 8; ++t) {
            afrag[t]     = f2bf(Asrc[m * 32 + kb + t]);
            afrag[8 + t] = f2bf(Asrc[m * 32 + kb + 16 + t]);
        }
        v16bf bfrag = *(const v16bf*)(bptr + (size_t)kt * 512);
        acc = __builtin_amdgcn_wmma_f32_16x16x32_bf16(
            false, afrag, false, bfrag, (short)0, acc, false, false);
        __syncthreads();                 // buffer kt free before re-issue at kt+2
    }
    int n = lane & 15;
    int mbase = (lane >> 4) * 8;
    float bias = b1[wave * 16 + n];
#pragma unroll
    for (int v = 0; v < 8; ++v) {
        int node = row0 + mbase + v;
        float val = acc[v] + bias;
        x1[(size_t)node * HIDD + wave * 16 + n] = val > 0.0f ? val : 0.0f;
    }
}

// ------------------------------------------- shared A-tile staging (layer 2)
// Convert a 16x128 fp32 x1 tile into bf16 fragment layout Af[kt][lane][16].

__device__ __forceinline__ void stage_x1_tile(const float* __restrict__ x1,
                                              int row0, int tid, int nthreads,
                                              __bf16 (*Af)[32][16]) {
    for (int i = tid; i < 16 * 64; i += nthreads) {   // 1024 float2 pairs
        int m = i & 15;
        int p = i >> 4;                // pair index 0..63
        int k = 2 * p;
        float2 v = *(const float2*)(x1 + (size_t)(row0 + m) * HIDD + k);
        int lane2 = m + ((k >> 3) & 1) * 16;
        int t2    = (k & 7) + ((k >> 4) & 1) * 8;
        int kt    = k >> 5;
        v2bf pk;
        pk[0] = f2bf(v.x);
        pk[1] = f2bf(v.y);
        *(v2bf*)&Af[kt][lane2][t2] = pk;
    }
}

// --------------------------------------------------------------- GEMM 2
// H2[n, r*64+j] = sum_k x1[n,k] * W2[r,k,j]; K = 128 (4 tiles), 1024 cols.

__global__ __launch_bounds__(256) void gemm2_kernel(
    const float* __restrict__ x1,     // [N, 128]
    const __bf16* __restrict__ B2pk,  // [64][4][32][16]
    float* __restrict__ H2) {         // [N, 1024]
    __shared__ __bf16 Af[4][32][16];  // bf16 fragment-layout A tile (4 KB)
    int row0 = blockIdx.x * 16;
    int wave = threadIdx.x >> 5;
    int lane = threadIdx.x & 31;
    int ct   = blockIdx.y * 8 + wave;                 // 0..63

    stage_x1_tile(x1, row0, threadIdx.x, 256, Af);
    __syncthreads();

    const __bf16* bptr = B2pk + ((size_t)ct * 4) * 512 + lane * 16;
    v8f acc = {};
#pragma unroll
    for (int kt = 0; kt < 4; ++kt) {
        v16bf afrag = *(const v16bf*)&Af[kt][lane][0];
        v16bf bfrag = *(const v16bf*)(bptr + (size_t)kt * 512);
        acc = __builtin_amdgcn_wmma_f32_16x16x32_bf16(
            false, afrag, false, bfrag, (short)0, acc, false, false);
    }
    int n = lane & 15;
    int mbase = (lane >> 4) * 8;
#pragma unroll
    for (int v = 0; v < 8; ++v) {
        int node = row0 + mbase + v;
        H2[(size_t)node * (RR * EMBD) + ct * 16 + n] = acc[v];
    }
}

// ------------------------------------------------------------ root2 GEMM
// out[n, c] = sum_k x1[n,k] * root2[k,c] + b2[c]   (initializes d_out)

__global__ __launch_bounds__(128) void gemm_root2_kernel(
    const float* __restrict__ x1,     // [N, 128]
    const __bf16* __restrict__ R2pk,  // [4][4][32][16]
    const float* __restrict__ b2,     // [64]
    float* __restrict__ out) {        // [N, 64]
    __shared__ __bf16 Af[4][32][16];
    int row0 = blockIdx.x * 16;
    int wave = threadIdx.x >> 5;                      // 0..3 -> col tile
    int lane = threadIdx.x & 31;

    stage_x1_tile(x1, row0, threadIdx.x, 128, Af);
    __syncthreads();

    const __bf16* bptr = R2pk + ((size_t)wave * 4) * 512 + lane * 16;
    v8f acc = {};
#pragma unroll
    for (int kt = 0; kt < 4; ++kt) {
        v16bf afrag = *(const v16bf*)&Af[kt][lane][0];
        v16bf bfrag = *(const v16bf*)(bptr + (size_t)kt * 512);
        acc = __builtin_amdgcn_wmma_f32_16x16x32_bf16(
            false, afrag, false, bfrag, (short)0, acc, false, false);
    }
    int n = lane & 15;
    int mbase = (lane >> 4) * 8;
    float bias = b2[wave * 16 + n];
#pragma unroll
    for (int v = 0; v < 8; ++v) {
        int node = row0 + mbase + v;
        out[(size_t)node * EMBD + wave * 16 + n] = acc[v] + bias;
    }
}

// ----------------------------------------------------------------- launch

extern "C" void kernel_launch(void* const* d_in, const int* in_sizes, int n_in,
                              void* d_out, int out_size, void* d_ws, size_t ws_size,
                              hipStream_t stream) {
    const int*   edge_index = (const int*)d_in[0];   // [2, E]
    const int*   edge_type  = (const int*)d_in[1];   // [E]
    const float* emb   = (const float*)d_in[2];      // [N, 64]
    const float* W1    = (const float*)d_in[3];      // [R, 64, 128]
    const float* root1 = (const float*)d_in[4];      // [64, 128]
    const float* b1    = (const float*)d_in[5];      // [128]
    const float* W2    = (const float*)d_in[6];      // [R, 128, 64]
    const float* root2 = (const float*)d_in[7];      // [128, 64]
    const float* b2    = (const float*)d_in[8];      // [64]
    float* out = (float*)d_out;

    char* ws = (char*)d_ws;
    size_t off = 0;
    float* wseg = (float*)(ws + off); off += (size_t)NODES * RR * 4;       // 3.2 MB
    float* x1   = (float*)(ws + off); off += (size_t)NODES * HIDD * 4;     // 25.6 MB
    float* big  = (float*)(ws + off); off += (size_t)NODES * RR * EMBD * 4;// 204.8 MB
    __bf16* B1pk = (__bf16*)(ws + off); off += (size_t)8 * KT1 * 512 * 2;  // 272 KB
    __bf16* B2pk = (__bf16*)(ws + off); off += (size_t)64 * 4 * 512 * 2;   // 256 KB
    __bf16* R2pk = (__bf16*)(ws + off); off += (size_t)4 * 4 * 512 * 2;    // 16 KB

    const int NT   = NODES / 16;              // 3125 node tiles (exact)
    const int segN = NODES * RR;              // 800000
    const int bigN4 = NODES * RR * EMBD / 4;  // 12.8M float4
    const int edgeWaves = (NEDGE + 7) / 8;

    // pack weights into fragment-ready bf16 (L2-resident, done once per call)
    pack_b_kernel<<<(1024 * 128 + 255) / 256, 256, 0, stream>>>(
        W1, B1pk, 1024, 128, 128, 64, KT1, 0, 5);
    pack_b_kernel<<<(64 * 128 + 255) / 256, 256, 0, stream>>>(
        root1, B1pk, 64, 128, 128, 64, KT1, 32, 1);
    pack_b_kernel<<<(128 * 1024 + 255) / 256, 256, 0, stream>>>(
        W2, B2pk, 128, 1024, 64, 8192, 4, 0, 2);
    pack_b_kernel<<<(128 * 64 + 255) / 256, 256, 0, stream>>>(
        root2, R2pk, 128, 64, 64, 64, 4, 0, 2);

    // mean-normalization weights (shared by both layers)
    fill_zero4_kernel<<<(segN / 4 + 255) / 256, 256, 0, stream>>>((float4*)wseg, segN / 4);
    count_kernel<<<(NEDGE + 255) / 256, 256, 0, stream>>>(edge_index, edge_type, wseg);
    invert_kernel<<<(segN + 255) / 256, 256, 0, stream>>>(wseg, segN);

    // layer 1: aggregate -> big GEMM (relations + root fused into K=1088)
    fill_zero4_kernel<<<(bigN4 + 255) / 256, 256, 0, stream>>>((float4*)big, bigN4);
    scatter1_kernel<<<edgeWaves, 256, 0, stream>>>(edge_index, edge_type, emb, wseg, big);
    gemm1_kernel<<<NT, 256, 0, stream>>>(big, emb, B1pk, b1, x1);

    // layer 2: transform (all relations in one GEMM) -> scatter
    gemm2_kernel<<<dim3(NT, 8), 256, 0, stream>>>(x1, B2pk, big);
    gemm_root2_kernel<<<NT, 128, 0, stream>>>(x1, R2pk, b2, out);
    scatter2_kernel<<<edgeWaves, 256, 0, stream>>>(edge_index, edge_type, big, wseg, out);
}
